// PredicateUnit_34849364640337
// MI455X (gfx1250) — matrix operand order repaired and checked by reading, not compiled, not run on hardware
//
#include <hip/hip_runtime.h>
#include <hip/hip_bf16.h>

#define IMG_C   3
#define IMG_H   64
#define IMG_W   96
#define N_OBJ   5
#define BATCH   64
#define PRED    4
#define CONV_OUT 32
#define OUTC    (PRED * CONV_OUT)      // 128 output channels
#define OUT_H   32
#define OUT_W   48
#define SPATIAL (OUT_H * OUT_W)        // 1536
#define KDIM    54                     // 6*3*3
#define KPAD    64
#define TUPLES  (BATCH * N_OBJ * N_OBJ) // 1600
#define IMGS    (2 * TUPLES)            // 3200 (state + state_next)
#define MTILE   128
#define MTILES  (SPATIAL / MTILE)       // 12
#define LDA     72                      // padded LDS row stride in halves (16B aligned, conflict-avoiding)

typedef __attribute__((ext_vector_type(16))) _Float16 v16h;
typedef __attribute__((ext_vector_type(8)))  _Float16 v8h;
typedef __attribute__((ext_vector_type(8)))  float    v8f;

// ---------------------------------------------------------------------------
// Kernel 1: pack conv weights (128 x 54, f32) -> f16 [col][k], K padded to 64
// ---------------------------------------------------------------------------
__global__ void init_pack(const float* __restrict__ conv_w, _Float16* __restrict__ pw) {
    int idx = blockIdx.x * 256 + threadIdx.x;
    if (idx < OUTC * KPAD) {
        int col = idx >> 6;
        int k   = idx & 63;
        float v = (k < KDIM) ? conv_w[col * KDIM + k] : 0.0f;
        pw[idx] = (_Float16)v;
    }
}

// ---------------------------------------------------------------------------
// Kernel 2: implicit-GEMM conv (stride-2 SAME 3x3, 6->128ch) on WMMA f16,
// fused bias + ReLU + spatial-sum epilogue. One block per tuple image.
// GEMM: M = 1536 spatial positions, N = 128 channels, K = 54 (padded 64).
// ---------------------------------------------------------------------------
__global__ void __launch_bounds__(256, 2)
conv_wmma(const float* __restrict__ state, const float* __restrict__ state_next,
          const _Float16* __restrict__ pw, const float* __restrict__ conv_b,
          float* __restrict__ sums) {
    __shared__ __align__(16) _Float16 lA[MTILE * LDA];   // im2col A tile (128 x 64 f16)
    __shared__ __align__(16) _Float16 lW[OUTC * LDA];    // B (weights)  (128 cols x 64 K)
    __shared__ float lpart[OUTC * 16];                   // deterministic reduction buffer

    const int tid  = threadIdx.x;
    const int img  = blockIdx.x;

    const float* src = (img < TUPLES) ? state : state_next;
    const int t  = img % TUPLES;
    const int b  = t / (N_OBJ * N_OBJ);
    const int pr = t % (N_OBJ * N_OBJ);
    const int oi = pr / N_OBJ;
    const int oj = pr % N_OBJ;
    const float* __restrict__ srcA = src + ((size_t)b * N_OBJ + oi) * (IMG_C * IMG_H * IMG_W);
    const float* __restrict__ srcB = src + ((size_t)b * N_OBJ + oj) * (IMG_C * IMG_H * IMG_W);

    // ---- load packed weights to LDS (dword copies, once per block) ----
    const uint32_t* pw32 = (const uint32_t*)pw;
    #pragma unroll
    for (int i = 0; i < (OUTC * KPAD / 2) / 256; ++i) {   // 4096 dwords, 16 iters
        int idx = i * 256 + tid;
        int col = idx >> 5;          // 32 dwords per column
        int kd  = idx & 31;
        *reinterpret_cast<uint32_t*>(&lW[col * LDA + kd * 2]) = pw32[idx];
    }

    const int wave = tid >> 5;
    const int lane = tid & 31;
    const int hl   = (lane < 16) ? 0 : 1;   // lane half per ISA A/B layouts
    const int lr   = lane & 15;
    const int arow = wave * 16 + lr;

    float sacc[8];
    #pragma unroll
    for (int nt = 0; nt < 8; ++nt) sacc[nt] = 0.0f;

    for (int mtile = 0; mtile < MTILES; ++mtile) {
        __syncthreads();   // lA reuse guard (also covers first-use of lW)

        // ---- cooperative im2col: 128 rows x 64 K, f32 -> f16 into LDS ----
        #pragma unroll 4
        for (int i = 0; i < (MTILE * KPAD) / 256; ++i) {  // 32 iters
            int idx = i * 256 + tid;
            int r = idx & (MTILE - 1);
            int k = idx >> 7;
            int m  = mtile * MTILE + r;
            int oy = m / OUT_W, ox = m % OUT_W;
            float v = 0.0f;
            if (k < KDIM) {
                int c = k / 9, q = k % 9;
                int ky = q / 3, kx = q % 3;
                int iy = 2 * oy + ky;       // SAME pad: pad_begin = 0, pad_end = 1
                int ix = 2 * ox + kx;
                if (iy < IMG_H && ix < IMG_W) {
                    int off = iy * IMG_W + ix;
                    v = (c < IMG_C) ? srcA[c * (IMG_H * IMG_W) + off]
                                    : srcB[(c - IMG_C) * (IMG_H * IMG_W) + off];
                }
            }
            lA[r * LDA + k] = (_Float16)v;
        }
        __syncthreads();

        // ---- A fragments (16-bit A layout: lanes<16 K{0-7,16-23}, lanes>=16 K{8-15,24-31}) ----
        v16h afrag[2];
        #pragma unroll
        for (int ks = 0; ks < 2; ++ks) {
            int k0 = ks * 32 + hl * 8;
            v8h lo = *reinterpret_cast<const v8h*>(&lA[arow * LDA + k0]);
            v8h hi = *reinterpret_cast<const v8h*>(&lA[arow * LDA + k0 + 16]);
            #pragma unroll
            for (int e = 0; e < 8; ++e) { afrag[ks][e] = lo[e]; afrag[ks][8 + e] = hi[e]; }
        }

        // ---- 8 N-tiles x 2 K-steps of v_wmma_f32_16x16x32_f16, fused epilogue ----
        #pragma unroll
        for (int nt = 0; nt < 8; ++nt) {
            int col = nt * 16 + lr;
            v8f acc = {};
            #pragma unroll
            for (int ks = 0; ks < 2; ++ks) {
                // B layout: lane = column; lanes<16 hold K{0-7,16-23}, lanes>=16 hold K{8-15,24-31}
                int k0 = ks * 32 + hl * 8;
                v8h blo = *reinterpret_cast<const v8h*>(&lW[col * LDA + k0]);
                v8h bhi = *reinterpret_cast<const v8h*>(&lW[col * LDA + k0 + 16]);
                v16h bfrag;
                #pragma unroll
                for (int e = 0; e < 8; ++e) { bfrag[e] = blo[e]; bfrag[8 + e] = bhi[e]; }
                acc = __builtin_amdgcn_wmma_f32_16x16x32_f16(
                          false, afrag[ks], false, bfrag,
                          (short)0, acc, false, false);
            }
            // bias + ReLU + sum over this lane's 8 rows (C/D layout: vgpr r -> row r or r+8)
            float bias = conv_b[col];
            float s = 0.0f;
            #pragma unroll
            for (int r = 0; r < 8; ++r) {
                float vv = acc[r] + bias;
                s += (vv > 0.0f) ? vv : 0.0f;
            }
            sacc[nt] += s;
        }
    }

    // ---- deterministic fixed-order reduction (no float atomics) ----
    #pragma unroll
    for (int nt = 0; nt < 8; ++nt) {
        int col = nt * 16 + lr;
        lpart[col * 16 + wave * 2 + hl] = sacc[nt];
    }
    __syncthreads();
    if (tid < OUTC) {
        float s = 0.0f;
        #pragma unroll
        for (int i = 0; i < 16; ++i) s += lpart[tid * 16 + i];
        sums[(size_t)img * OUTC + tid] = s;   // plain store: unique writer per (img, ch)
    }
}

// ---------------------------------------------------------------------------
// Kernel 3: mean, 128->4 head, sigmoid(logits/temp)
// ---------------------------------------------------------------------------
__global__ void finish_kernel(const float* __restrict__ sums, const float* __restrict__ w2,
                              const float* __restrict__ b2, const float* __restrict__ temp,
                              float* __restrict__ out) {
    int idx = blockIdx.x * 256 + threadIdx.x;   // (img, p)
    if (idx >= IMGS * PRED) return;
    int img = idx >> 2;
    int p   = idx & 3;
    const float inv = 1.0f / (float)SPATIAL;
    float acc = b2[p];
    #pragma unroll
    for (int c = 0; c < CONV_OUT; ++c)
        acc += (sums[(size_t)img * OUTC + p * CONV_OUT + c] * inv) * w2[p * CONV_OUT + c];
    float lg = acc / temp[0];
    float pred = 1.0f / (1.0f + __expf(-lg));
    int t = img % TUPLES;
    float* dst = (img < TUPLES) ? out : (out + TUPLES * PRED);
    dst[t * PRED + p] = pred;
}

// ---------------------------------------------------------------------------
// Kernel 4: adjacency mask adj[b, t] = (t // 25 == b)
// ---------------------------------------------------------------------------
__global__ void adj_kernel(float* __restrict__ out) {
    int idx = blockIdx.x * 256 + threadIdx.x;
    if (idx >= BATCH * TUPLES) return;
    int bb = idx / TUPLES;
    int t  = idx % TUPLES;
    out[2 * TUPLES * PRED + idx] = (t / (N_OBJ * N_OBJ) == bb) ? 1.0f : 0.0f;
}

// ---------------------------------------------------------------------------
extern "C" void kernel_launch(void* const* d_in, const int* in_sizes, int n_in,
                              void* d_out, int out_size, void* d_ws, size_t ws_size,
                              hipStream_t stream) {
    const float* state      = (const float*)d_in[0];
    const float* state_next = (const float*)d_in[1];
    const float* conv_w     = (const float*)d_in[2];
    const float* conv_b     = (const float*)d_in[3];
    const float* w2         = (const float*)d_in[4];
    const float* b2         = (const float*)d_in[5];
    /* d_in[6] = n_obj (always N_OBJ, shapes are static) */
    const float* temp       = (const float*)d_in[7];
    float* out = (float*)d_out;

    // workspace: [0, IMGS*OUTC*4) per-image channel sums; then packed f16 weights
    float*    sums = (float*)d_ws;
    _Float16* pw   = (_Float16*)((char*)d_ws + (size_t)IMGS * OUTC * sizeof(float));

    init_pack<<<(OUTC * KPAD + 255) / 256, 256, 0, stream>>>(conv_w, pw);
    conv_wmma<<<IMGS, 256, 0, stream>>>(state, state_next, pw, conv_b, sums);
    finish_kernel<<<(IMGS * PRED + 255) / 256, 256, 0, stream>>>(sums, w2, b2, temp, out);
    adj_kernel<<<(BATCH * TUPLES + 255) / 256, 256, 0, stream>>>(out);
}
